// Decoder_88132728914386
// MI455X (gfx1250) — matrix operand order repaired and checked by reading, not compiled, not run on hardware
//
#include <hip/hip_runtime.h>
#include <hip/hip_bf16.h>
#include <math.h>

// ---------------------------------------------------------------- types
typedef __bf16 bf16;
typedef __bf16 bf16x8  __attribute__((ext_vector_type(8)));
typedef __bf16 bf16x16 __attribute__((ext_vector_type(16)));
typedef float  v8f     __attribute__((ext_vector_type(8)));
typedef int    v4i     __attribute__((ext_vector_type(4)));

#if defined(__has_builtin)
#if __has_builtin(__builtin_amdgcn_global_load_async_to_lds_b128)
#define HAVE_ASYNC 1
#else
#define HAVE_ASYNC 0
#endif
#else
#define HAVE_ASYNC 0
#endif

// ---------------------------------------------------------------- problem dims
constexpr int BN   = 8;
constexpr int S    = 224;
constexpr int CIN  = 256;
constexpr int COUT = 16;
constexpr int TOPK = 200;
constexpr int MTOT = BN * S * S;          // 401408 output pixels
constexpr int KTOT = 9 * CIN;             // 2304 (3x3 conv as GEMM K)
constexpr int NK   = KTOT / 32;           // 72 K-steps

// ---------------------------------------------------------------- output offsets (flat fp32, reference tuple order)
constexpr size_t NPIX      = (size_t)MTOT;
constexpr size_t O_CENTER  = 0;
constexpr size_t O_DISP    = O_CENTER  + NPIX;
constexpr size_t O_CPTS    = O_DISP    + NPIX * 4;
constexpr size_t O_CSC     = O_CPTS    + (size_t)BN * TOPK * 2;
constexpr size_t O_DISPACT = O_CSC     + (size_t)BN * TOPK;
constexpr size_t O_LINE    = O_DISPACT + NPIX * 4;
constexpr size_t O_CORNER  = O_LINE    + NPIX;
constexpr size_t O_KPTS    = O_CORNER  + NPIX;
constexpr size_t O_KSC     = O_KPTS    + (size_t)BN * TOPK * 2;
constexpr size_t O_ORGC    = O_KSC     + (size_t)BN * TOPK;
constexpr size_t O_ORGD    = O_ORGC    + NPIX;
constexpr size_t O_OPTS    = O_ORGD    + NPIX * 4;
constexpr size_t O_OSC     = O_OPTS    + (size_t)BN * TOPK * 2;
constexpr size_t O_ODIST   = O_OSC     + (size_t)BN * TOPK;
constexpr size_t O_ODEG    = O_ODIST   + NPIX;
constexpr size_t O_SDIST   = O_ODEG    + NPIX;
constexpr size_t O_SDEG    = O_SDIST   + NPIX;

// ---------------------------------------------------------------- helpers
__device__ __forceinline__ bf16x16 cat16(bf16x8 lo, bf16x8 hi) {
    return __builtin_shufflevector(lo, hi, 0,1,2,3,4,5,6,7,8,9,10,11,12,13,14,15);
}
__device__ __forceinline__ bf16x8 cvt8(float4 a, float4 b) {
    bf16x8 r;
    r[0]=(bf16)a.x; r[1]=(bf16)a.y; r[2]=(bf16)a.z; r[3]=(bf16)a.w;
    r[4]=(bf16)b.x; r[5]=(bf16)b.y; r[6]=(bf16)b.z; r[7]=(bf16)b.w;
    return r;
}
__device__ __forceinline__ void async_cp16(const bf16* g, bf16* l) {
#if HAVE_ASYNC
    typedef __attribute__((address_space(1))) v4i gv4i;
    typedef __attribute__((address_space(3))) v4i lv4i;
    v4i* gp = (v4i*)const_cast<bf16*>(g);
    v4i* lp = (v4i*)l;
    __builtin_amdgcn_global_load_async_to_lds_b128((gv4i*)gp, (lv4i*)lp, 0, 0);
#else
    (void)g; (void)l;
#endif
}
template <int N> __device__ __forceinline__ void wait_async() {
#if HAVE_ASYNC
#if __has_builtin(__builtin_amdgcn_s_wait_asynccnt)
    __builtin_amdgcn_s_wait_asynccnt(N);
#else
    asm volatile("s_wait_asynccnt %0" :: "i"(N) : "memory");
#endif
#endif
}

// ---------------------------------------------------------------- BN folding + zero page
__global__ void bn_prep(const float* __restrict__ g0, const float* __restrict__ b0,
                        const float* __restrict__ m0, const float* __restrict__ v0,
                        const float* __restrict__ g1, const float* __restrict__ b1,
                        const float* __restrict__ m1, const float* __restrict__ v1,
                        float* __restrict__ sc0, float* __restrict__ sh0,
                        float* __restrict__ sc1, float* __restrict__ sh1,
                        float* __restrict__ zp) {
    int i = threadIdx.x;
    float s0 = g0[i] * rsqrtf(v0[i] + 1e-5f);
    sc0[i] = s0; sh0[i] = b0[i] - m0[i] * s0;
    float s1 = g1[i] * rsqrtf(v1[i] + 1e-5f);
    sc1[i] = s1; sh1[i] = b1[i] - m1[i] * s1;
    if (i < 64) zp[i] = 0.f;
}

// ---------------------------------------------------------------- x fp32 -> bf16 prolog (8 elems/thread)
__global__ void cvt_x_bf16(const float* __restrict__ x, bf16* __restrict__ xb) {
    const size_t i8 = ((size_t)blockIdx.x * 256 + threadIdx.x) * 8;
    const float4* p = reinterpret_cast<const float4*>(x + i8);
    bf16x8 r = cvt8(p[0], p[1]);
    *reinterpret_cast<uint4*>(xb + i8) = __builtin_bit_cast(uint4, r);
}

// ---------------------------------------------------------------- weight pack
__global__ void pack_w3x3(const float* __restrict__ w, bf16* __restrict__ wp) {
    int i = blockIdx.x * 256 + threadIdx.x;          // 9*CIN*CIN
    if (i >= 9 * CIN * CIN) return;
    int n  = i % CIN;
    int tc = i / CIN;                                // t*CIN + c
    wp[(size_t)n * KTOT + tc] = (bf16)w[(size_t)tc * CIN + n];
}
__global__ void pack_w1x1(const float* __restrict__ w, bf16* __restrict__ wp) {
    int i = blockIdx.x * 256 + threadIdx.x;          // CIN*COUT
    if (i >= CIN * COUT) return;
    int n = i % COUT, c = i / COUT;
    wp[(size_t)n * CIN + c] = (bf16)w[i];
}

// ---------------------------------------------------------------- 3x3 conv, implicit GEMM, bf16 WMMA
// BLOCK: M=128, N=256, K-step 32. 8 waves = 4(M) x 2(N); wave tile 32 x 128 = 16 WMMA.
// Double-buffered LDS; staging via GLOBAL_LOAD_ASYNC_TO_LDS_B128 when available.
__global__ __launch_bounds__(256) void conv3x3_wmma(
        const bf16* __restrict__ in,         // (B,S,S,CIN) bf16
        const bf16* __restrict__ wp,         // (n)(KTOT) N-major
        const float* __restrict__ scale,
        const float* __restrict__ shift,
        const float* __restrict__ zpage,     // 256B of zeros
        bf16* __restrict__ outp)             // (B,S,S,CIN) bf16
{
    __shared__ __align__(16) bf16 Alds[2][128][40];   // 32 K + 8 pad halves
    __shared__ __align__(16) bf16 Blds[2][256][40];   // [n][k]

    const int tid  = threadIdx.x;
    const int lane = tid & 31;
    const int wv   = tid >> 5;
    const int lm   = lane & 15;
    const int lg   = lane >> 4;
    const int wm   = wv >> 1;        // 0..3  (M: 32 rows each)
    const int wn   = wv & 1;         // 0..1  (N: 128 cols each)

    const int m0 = blockIdx.x * 128;

    // staging task: A row r = tid/2, half = tid&1; B row = tid
    const int r    = tid >> 1;
    const int half = tid & 1;
    const int m    = m0 + r;
    const int bimg = m / (S * S);
    const int yx   = m % (S * S);
    const int y    = yx / S;
    const int x    = yx % S;
    const bf16* zp = reinterpret_cast<const bf16*>(zpage);

    v8f acc[2][8];
    #pragma unroll
    for (int mi = 0; mi < 2; ++mi)
        #pragma unroll
        for (int ni = 0; ni < 8; ++ni)
            acc[mi][ni] = (v8f)0.f;

    // ---- one staging step (A: 2x16B per thread, B: 4x16B per thread)
    auto stage = [&](int ks, int buf) {
        const int kk = ks * 32;
        const int t  = kk >> 8;                  // tap 0..8
        const int iy = y + (t / 3) - 1;
        const int ix = x + (t % 3) - 1;
        const int c0 = (kk & 255) + half * 16;
        const bool ok = (iy >= 0 && iy < S && ix >= 0 && ix < S);
        const size_t abase = (((size_t)bimg * S + iy) * S + ix) * CIN + c0;
        const bf16* wrow = wp + (size_t)tid * KTOT + kk;
#if HAVE_ASYNC
        const bf16* ga0 = ok ? (in + abase)     : zp;
        const bf16* ga1 = ok ? (in + abase + 8) : zp;
        async_cp16(ga0, &Alds[buf][r][half * 16]);
        async_cp16(ga1, &Alds[buf][r][half * 16 + 8]);
        async_cp16(wrow,      &Blds[buf][tid][0]);
        async_cp16(wrow + 8,  &Blds[buf][tid][8]);
        async_cp16(wrow + 16, &Blds[buf][tid][16]);
        async_cp16(wrow + 24, &Blds[buf][tid][24]);
#else
        uint4 a0 = {0,0,0,0}, a1 = {0,0,0,0};
        if (ok) {
            const uint4* p = reinterpret_cast<const uint4*>(in + abase);
            a0 = p[0]; a1 = p[1];
        }
        *reinterpret_cast<uint4*>(&Alds[buf][r][half * 16])     = a0;
        *reinterpret_cast<uint4*>(&Alds[buf][r][half * 16 + 8]) = a1;
        const uint4* q = reinterpret_cast<const uint4*>(wrow);
        *reinterpret_cast<uint4*>(&Blds[buf][tid][0])  = q[0];
        *reinterpret_cast<uint4*>(&Blds[buf][tid][8])  = q[1];
        *reinterpret_cast<uint4*>(&Blds[buf][tid][16]) = q[2];
        *reinterpret_cast<uint4*>(&Blds[buf][tid][24]) = q[3];
        (void)zp;
#endif
    };

    stage(0, 0);
    for (int ks = 0; ks < NK; ++ks) {
        const int buf = ks & 1;
        if (ks + 1 < NK) {
            stage(ks + 1, buf ^ 1);
            wait_async<6>();     // previous stage's 6 async ops completed (in-order)
        } else {
            wait_async<0>();
        }
        __syncthreads();

        bf16x16 af[2];
        #pragma unroll
        for (int mi = 0; mi < 2; ++mi) {
            const int ar = wm * 32 + mi * 16 + lm;
            bf16x8 lo = *reinterpret_cast<const bf16x8*>(&Alds[buf][ar][lg * 8]);
            bf16x8 hi = *reinterpret_cast<const bf16x8*>(&Alds[buf][ar][16 + lg * 8]);
            af[mi] = cat16(lo, hi);
        }
        #pragma unroll
        for (int ni = 0; ni < 8; ++ni) {
            const int br = wn * 128 + ni * 16 + lm;
            bf16x8 lo = *reinterpret_cast<const bf16x8*>(&Blds[buf][br][lg * 8]);
            bf16x8 hi = *reinterpret_cast<const bf16x8*>(&Blds[buf][br][16 + lg * 8]);
            bf16x16 bfr = cat16(lo, hi);
            acc[0][ni] = __builtin_amdgcn_wmma_f32_16x16x32_bf16(
                false, af[0], false, bfr, (short)0, acc[0][ni], false, false);
            acc[1][ni] = __builtin_amdgcn_wmma_f32_16x16x32_bf16(
                false, af[1], false, bfr, (short)0, acc[1][ni], false, false);
        }
        __syncthreads();   // all waves done reading before next overwrite of this buffer
    }

    // ---- epilogue: BN + ReLU, bf16 NHWC
    #pragma unroll
    for (int ni = 0; ni < 8; ++ni) {
        const int n = wn * 128 + ni * 16 + lm;
        const float sc = scale[n], sh = shift[n];
        #pragma unroll
        for (int mi = 0; mi < 2; ++mi) {
            #pragma unroll
            for (int v = 0; v < 8; ++v) {
                const int mm = m0 + wm * 32 + mi * 16 + lg * 8 + v;
                float val = acc[mi][ni][v] * sc + sh;
                val = fmaxf(val, 0.f);
                outp[(size_t)mm * CIN + n] = (bf16)val;
            }
        }
    }
}

// ---------------------------------------------------------------- 1x1 conv (K=256, N=16) -> out_map fp32
__global__ __launch_bounds__(256) void conv1x1_wmma(
        const bf16* __restrict__ h,          // (M, 256) bf16
        const bf16* __restrict__ wp2,        // (16 n)(256 k)
        const float* __restrict__ bias,
        float* __restrict__ om)              // (M, 16)
{
    const int tid  = threadIdx.x;
    const int lane = tid & 31;
    const int wv   = tid >> 5;
    const int lm   = lane & 15;
    const int lg   = lane >> 4;
    const int m0   = blockIdx.x * 128 + wv * 16;

    v8f acc = (v8f)0.f;
    #pragma unroll
    for (int ks = 0; ks < 8; ++ks) {
        const int kk = ks * 32;
        const size_t ab = (size_t)(m0 + lm) * CIN + kk;
        bf16x8 alo = *reinterpret_cast<const bf16x8*>(h + ab + lg * 8);
        bf16x8 ahi = *reinterpret_cast<const bf16x8*>(h + ab + 16 + lg * 8);
        bf16x16 a = cat16(alo, ahi);
        const size_t bb = (size_t)lm * CIN + kk + lg * 8;
        bf16x8 blo = *reinterpret_cast<const bf16x8*>(wp2 + bb);
        bf16x8 bhi = *reinterpret_cast<const bf16x8*>(wp2 + bb + 16);
        bf16x16 b = cat16(blo, bhi);
        acc = __builtin_amdgcn_wmma_f32_16x16x32_bf16(
            false, a, false, b, (short)0, acc, false, false);
    }
    const float bv = bias[lm];
    #pragma unroll
    for (int v = 0; v < 8; ++v) {
        const int mm = m0 + lg * 8 + v;
        om[(size_t)mm * COUT + lm] = acc[v] + bv;
    }
}

// ---------------------------------------------------------------- channel scatter + sigmoids
__device__ __forceinline__ float sigmoidf(float r) { return 1.f / (1.f + __expf(-r)); }

__global__ void postproc_maps(const float* __restrict__ om, float* __restrict__ out) {
    const int p = blockIdx.x * 256 + threadIdx.x;
    if (p >= MTOT) return;
    const float* v = om + (size_t)p * COUT;
    out[O_CENTER + p] = v[0];
    out[O_LINE   + p] = v[1];
    out[O_CORNER + p] = v[2];
    #pragma unroll
    for (int i = 0; i < 4; ++i) {
        const float d = v[3 + i];
        out[O_DISP    + (size_t)p * 4 + i] = d;
        out[O_DISPACT + (size_t)p * 4 + i] = d;
        out[O_ORGD    + (size_t)p * 4 + i] = v[8 + i];
    }
    out[O_ORGC  + p] = v[7];
    out[O_ODIST + p] = sigmoidf(v[12]);
    out[O_ODEG  + p] = sigmoidf(v[13]);
    out[O_SDIST + p] = sigmoidf(v[14]);
    out[O_SDEG  + p] = sigmoidf(v[15]);
}

// ---------------------------------------------------------------- peak map (sigmoid monotone => NMS on raw logits)
__global__ void peaks_kernel(const float* __restrict__ om, float* __restrict__ peaks) {
    const int idx = blockIdx.x * 256 + threadIdx.x;   // 3 * MTOT
    if (idx >= 3 * MTOT) return;
    const int ci = idx / MTOT;
    const int p  = idx % MTOT;
    const int ch = (ci == 0) ? 0 : (ci == 1) ? 2 : 7;
    const int b  = p / (S * S);
    const int yx = p % (S * S);
    const int y  = yx / S, x = yx % S;
    const float r = om[(size_t)p * COUT + ch];
    float mx = r;
    #pragma unroll
    for (int dy = -1; dy <= 1; ++dy)
        #pragma unroll
        for (int dx = -1; dx <= 1; ++dx) {
            const int iy = y + dy, ix = x + dx;
            if (iy >= 0 && iy < S && ix >= 0 && ix < S)
                mx = fmaxf(mx, om[((((size_t)b * S + iy) * S + ix) * COUT) + ch]);
        }
    peaks[idx] = (r == mx) ? sigmoidf(r) : 0.f;
}

// ---------------------------------------------------------------- per-(channel,image) iterative top-200
__global__ __launch_bounds__(256) void topk_kernel(float* __restrict__ peaks,
                                                   float* __restrict__ out) {
    const int ci = blockIdx.x / BN;
    const int b  = blockIdx.x % BN;
    float* buf = peaks + (size_t)ci * MTOT + (size_t)b * (S * S);

    size_t o_pts, o_sc;
    if (ci == 0)      { o_pts = O_CPTS; o_sc = O_CSC; }
    else if (ci == 1) { o_pts = O_KPTS; o_sc = O_KSC; }
    else              { o_pts = O_OPTS; o_sc = O_OSC; }

    __shared__ float sv[256];
    __shared__ int   si[256];
    const int tid = threadIdx.x;

    for (int k = 0; k < TOPK; ++k) {
        float best = -1.f;
        int   bi   = 0x7fffffff;
        for (int i = tid; i < S * S; i += 256) {
            const float val = buf[i];
            if (val > best) { best = val; bi = i; }
        }
        sv[tid] = best; si[tid] = bi;
        __syncthreads();
        for (int off = 128; off > 0; off >>= 1) {
            if (tid < off) {
                const float v2 = sv[tid + off];
                const int   i2 = si[tid + off];
                if (v2 > sv[tid] || (v2 == sv[tid] && i2 < si[tid])) {
                    sv[tid] = v2; si[tid] = i2;
                }
            }
            __syncthreads();
        }
        const int   win = si[0];
        const float wv  = sv[0];
        if (tid == (win & 255)) buf[win] = -__builtin_inff();
        if (tid == 0) {
            out[o_sc + (size_t)b * TOPK + k] = wv;
            out[o_pts + ((size_t)b * TOPK + k) * 2 + 0] = (float)(win / S);
            out[o_pts + ((size_t)b * TOPK + k) * 2 + 1] = (float)(win % S);
        }
        __syncthreads();
    }
}

// ---------------------------------------------------------------- launch
extern "C" void kernel_launch(void* const* d_in, const int* in_sizes, int n_in,
                              void* d_out, int out_size, void* d_ws, size_t ws_size,
                              hipStream_t stream) {
    (void)in_sizes; (void)n_in; (void)out_size; (void)ws_size;

    const float* x  = (const float*)d_in[0];
    const float* w0 = (const float*)d_in[1];
    const float* g0 = (const float*)d_in[2];
    const float* b0 = (const float*)d_in[3];
    const float* m0 = (const float*)d_in[4];
    const float* v0 = (const float*)d_in[5];
    const float* w1 = (const float*)d_in[6];
    const float* g1 = (const float*)d_in[7];
    const float* b1 = (const float*)d_in[8];
    const float* m1 = (const float*)d_in[9];
    const float* v1 = (const float*)d_in[10];
    const float* w2 = (const float*)d_in[11];
    const float* b2 = (const float*)d_in[12];
    float* out = (float*)d_out;

    // workspace carve-up (256B aligned)
    char* base = (char*)d_ws;
    size_t off = 0;
    auto carve = [&](size_t bytes) {
        char* p = base + off;
        off += (bytes + 255) & ~(size_t)255;
        return p;
    };
    bf16*  xb    = (bf16*)carve((size_t)MTOT * CIN * 2);
    bf16*  h1    = (bf16*)carve((size_t)MTOT * CIN * 2);
    bf16*  h2    = (bf16*)carve((size_t)MTOT * CIN * 2);
    bf16*  wp0   = (bf16*)carve((size_t)CIN * KTOT * 2);
    bf16*  wp1   = (bf16*)carve((size_t)CIN * KTOT * 2);
    bf16*  wp2   = (bf16*)carve((size_t)COUT * CIN * 2);
    float* sc0   = (float*)carve(CIN * 4);
    float* sh0   = (float*)carve(CIN * 4);
    float* sc1   = (float*)carve(CIN * 4);
    float* sh1   = (float*)carve(CIN * 4);
    float* zp    = (float*)carve(256);
    float* om    = (float*)carve((size_t)MTOT * COUT * 4);
    float* peaks = (float*)carve((size_t)3 * MTOT * 4);

    bn_prep<<<1, 256, 0, stream>>>(g0, b0, m0, v0, g1, b1, m1, v1,
                                   sc0, sh0, sc1, sh1, zp);
    cvt_x_bf16<<<(int)(((size_t)MTOT * CIN / 8 + 255) / 256), 256, 0, stream>>>(x, xb);

    const int wpack_blocks = (9 * CIN * CIN + 255) / 256;
    pack_w3x3<<<wpack_blocks, 256, 0, stream>>>(w0, wp0);
    pack_w3x3<<<wpack_blocks, 256, 0, stream>>>(w1, wp1);
    pack_w1x1<<<(CIN * COUT + 255) / 256, 256, 0, stream>>>(w2, wp2);

    const int conv_blocks = MTOT / 128;   // 3136
    conv3x3_wmma<<<conv_blocks, 256, 0, stream>>>(xb, wp0, sc0, sh0, zp, h1);
    conv3x3_wmma<<<conv_blocks, 256, 0, stream>>>(h1, wp1, sc1, sh1, zp, h2);

    conv1x1_wmma<<<MTOT / 128, 256, 0, stream>>>(h2, wp2, b2, om);

    postproc_maps<<<(MTOT + 255) / 256, 256, 0, stream>>>(om, out);
    peaks_kernel<<<(3 * MTOT + 255) / 256, 256, 0, stream>>>(om, peaks);
    topk_kernel<<<3 * BN, 256, 0, stream>>>(peaks, out);
}
